// SwinTransformer_35253091565955
// MI455X (gfx1250) — matrix operand order repaired and checked by reading, not compiled, not run on hardware
//
#include <hip/hip_runtime.h>
#include <hip/hip_bf16.h>
#include <math.h>

typedef _Float16 h8  __attribute__((ext_vector_type(8)));
typedef _Float16 h16 __attribute__((ext_vector_type(16)));
typedef float    f8  __attribute__((ext_vector_type(8)));
typedef float    f4  __attribute__((ext_vector_type(4)));

union H16 { h16 v; h8 h[2]; };

__device__ __forceinline__ f8 f8zero() {
  f8 z = {0.f, 0.f, 0.f, 0.f, 0.f, 0.f, 0.f, 0.f};
  return z;
}
__device__ __forceinline__ h8 h8zero() {
  h8 z = {(_Float16)0, (_Float16)0, (_Float16)0, (_Float16)0,
          (_Float16)0, (_Float16)0, (_Float16)0, (_Float16)0};
  return z;
}

__device__ __forceinline__ f8 wmma16(h16 a, h16 b, f8 c) {
  return __builtin_amdgcn_wmma_f32_16x16x32_f16(false, a, false, b, (short)0, c,
                                                false, false);
}

// ---- async global->LDS staging (GLOBAL_LOAD_ASYNC_TO_LDS_B128, ASYNCcnt) ----
__device__ __forceinline__ void async_stage(const void* gsrc, const void* ldst,
                                            int chunks /* of 16B per 128 thr */,
                                            int tid) {
  unsigned lds0 = (unsigned)(unsigned long long)ldst;  // low 32b = LDS offset
  unsigned long long g0 = (unsigned long long)gsrc;
  for (int j = 0; j < chunks; j++) {
    unsigned lds = lds0 + (unsigned)(tid + 128 * j) * 16u;
    unsigned long long ga = g0 + (unsigned long long)(tid + 128 * j) * 16ull;
    asm volatile("global_load_async_to_lds_b128 %0, %1, off"
                 :: "v"(lds), "v"(ga) : "memory");
  }
  asm volatile("s_wait_asynccnt 0x0" ::: "memory");
}

// A fragment (16x32 f16): lane m=lane&15; lanes<16 hold K {k0..k0+7, k0+16..k0+23},
// lanes>=16 hold K {k0+8..k0+15, k0+24..k0+31}. Source row-major, stride in halfs.
__device__ __forceinline__ h16 loadA(const _Float16* base, int m0, int koff,
                                     int stride, int lane) {
  int m = lane & 15, hf = lane >> 4;
  const _Float16* p = base + (size_t)(m0 + m) * stride + koff + hf * 8;
  H16 r;
  r.h[0] = *(const h8*)p;
  r.h[1] = *(const h8*)(p + 16);
  return r.v;
}
// A fragment with logical K limited to 24 (head dim): zero the k>=24 chunk.
__device__ __forceinline__ h16 loadA_k24(const _Float16* base, int m0, int koff,
                                         int stride, int lane) {
  int m = lane & 15, hf = lane >> 4;
  const _Float16* p = base + (size_t)(m0 + m) * stride + koff + hf * 8;
  H16 r;
  r.h[0] = *(const h8*)p;                        // K 0..7 / 8..15 : valid
  r.h[1] = hf ? h8zero() : *(const h8*)(p + 16); // K 16..23 valid, 24..31 zero
  return r.v;
}
// B fragment (32x16 f16): lane n=lane&15 holds column n0+n; lanes<16 hold K {k0..k0+7, k0+16..k0+23}, lanes>=16 hold K {k0+8..k0+15, k0+24..k0+31}.
// Source stored as Bt[n][k] (row length = rowStride halfs).
__device__ __forceinline__ h16 loadB(const _Float16* base, int rowStride, int n0,
                                     int k0, int lane) {
  int n = lane & 15, hf = lane >> 4;
  const _Float16* p = base + (size_t)(n0 + n) * rowStride + k0 + hf * 8;
  H16 r;
  r.h[0] = *(const h8*)p;
  r.h[1] = *(const h8*)(p + 16);
  return r.v;
}
// B fragment with logical K limited to 24: lanes>=16 hi chunk covers K 24..31 -> zero.
__device__ __forceinline__ h16 loadB_k24(const _Float16* base, int rowStride, int n0,
                                         int koff, int lane) {
  int n = lane & 15, hf = lane >> 4;
  const _Float16* p = base + (size_t)(n0 + n) * rowStride + koff + hf * 8;
  H16 r;
  r.h[0] = *(const h8*)p;                        // K 0..7 / 8..15 : valid
  r.h[1] = hf ? h8zero() : *(const h8*)(p + 16); // K 16..23 valid, 24..31 zero
  return r.v;
}

// ---------------------------------------------------------------------------
// Kernel 0: weight transpose + f32->f16 convert. Wt[n][k] = W[k][n]
// ---------------------------------------------------------------------------
__global__ void prep_kernel(const float* __restrict__ wq, const float* __restrict__ wk,
                            const float* __restrict__ wv, const float* __restrict__ wo,
                            const float* __restrict__ w1, const float* __restrict__ w2,
                            _Float16* __restrict__ wqt, _Float16* __restrict__ wkt,
                            _Float16* __restrict__ wvt, _Float16* __restrict__ wot,
                            _Float16* __restrict__ w1t, _Float16* __restrict__ w2t) {
  int i = blockIdx.x * blockDim.x + threadIdx.x;
  int stride = gridDim.x * blockDim.x;
  for (int idx = i; idx < 36864; idx += stride) {
    if (idx < 9216) {  // 96x96 weights
      int n = idx / 96, k = idx % 96;
      wqt[n * 96 + k] = (_Float16)wq[k * 96 + n];
      wkt[n * 96 + k] = (_Float16)wk[k * 96 + n];
      wvt[n * 96 + k] = (_Float16)wv[k * 96 + n];
      wot[n * 96 + k] = (_Float16)wo[k * 96 + n];
    }
    {  // w1: [96,384] -> w1t [384][96]
      int n = idx / 96, k = idx % 96;
      w1t[n * 96 + k] = (_Float16)w1[k * 384 + n];
    }
    {  // w2: [384,96] -> w2t [96][384]
      int n = idx % 96, k = idx / 96;
      w2t[n * 384 + k] = (_Float16)w2[k * 96 + n];
    }
  }
}

// ---------------------------------------------------------------------------
// Kernel 1: LayerNorm1 + 4x4x4 window partition -> win_h f16 [4096][64][96]
// ---------------------------------------------------------------------------
__global__ __launch_bounds__(128) void ln1_kernel(const float* __restrict__ x,
                                                  const float* __restrict__ g,
                                                  const float* __restrict__ bta,
                                                  _Float16* __restrict__ winh) {
  size_t t = (size_t)blockIdx.x * blockDim.x + threadIdx.x;
  if (t >= 262144) return;
  const float* row = x + t * 96;
  float v[96];
#pragma unroll
  for (int cc = 0; cc < 24; cc++) *(f4*)(v + cc * 4) = *(const f4*)(row + cc * 4);
  float mu = 0.f;
#pragma unroll
  for (int c = 0; c < 96; c++) mu += v[c];
  mu *= (1.f / 96.f);
  float var = 0.f;
#pragma unroll
  for (int c = 0; c < 96; c++) {
    float d = v[c] - mu;
    var += d * d;
  }
  var *= (1.f / 96.f);
  float rs = rsqrtf(var + 1e-5f);
  int w = (int)(t & 63), hh = (int)((t >> 6) & 63), d = (int)((t >> 12) & 31),
      b = (int)(t >> 17);
  int wid = ((b * 8 + (d >> 2)) * 16 + (hh >> 2)) * 16 + (w >> 2);
  int l = ((d & 3) * 4 + (hh & 3)) * 4 + (w & 3);
  _Float16* dst = winh + ((size_t)wid * 64 + l) * 96;
#pragma unroll
  for (int cc = 0; cc < 12; cc++) {
    h8 pk;
#pragma unroll
    for (int j = 0; j < 8; j++) {
      int c = cc * 8 + j;
      pk[j] = (_Float16)((v[c] - mu) * rs * g[c] + bta[c]);
    }
    *(h8*)(dst + cc * 8) = pk;
  }
}

// ---------------------------------------------------------------------------
// Kernel 2: fused window attention (+residual, +LayerNorm2). 1 block / window.
// Each wave owns one 16-row (token) block: mt = wave (scalar), nt unrolled.
// Global-B GEMM loops are software-pipelined one tile deep.
// ---------------------------------------------------------------------------
__global__ __launch_bounds__(128) void attn_kernel(
    const float* __restrict__ x, const _Float16* __restrict__ winh,
    const _Float16* __restrict__ wqt, const float* __restrict__ bq,
    const _Float16* __restrict__ wkt, const float* __restrict__ bk,
    const _Float16* __restrict__ wvt, const float* __restrict__ bv,
    const _Float16* __restrict__ wot, const float* __restrict__ bo,
    const float* __restrict__ ln2g, const float* __restrict__ ln2b,
    float* __restrict__ x2, _Float16* __restrict__ ln2h) {
  __shared__ __align__(16) char smem[62976];
  _Float16* sQ = (_Float16*)(smem);             // 64 x 104 f16
  _Float16* sK = (_Float16*)(smem + 13312);     // 64 x 104 f16
  _Float16* sVt = (_Float16*)(smem + 26624);    // 96 x 72  f16 (V transposed)
  _Float16* sS = (_Float16*)(smem + 40448);     // 64 x 72  f16 (scores/attn)
  _Float16* sWin = (_Float16*)(smem + 49664);   // 64 x 96 f16 contiguous (12288 B)
  _Float16* sO = (_Float16*)(smem + 49664);     // reused: 64 x 104 f16 (13312 B)
  float* sX2 = (float*)(smem);                  // 64 x 100 f32 (reuses sQ/sK)

  const int tid = threadIdx.x;
  const int lane = tid & 31;
  const int mt = __builtin_amdgcn_readfirstlane(tid >> 5);  // scalar wave id
  const int nidx = lane & 15;
  const int mrow = (lane >> 4) * 8;
  const int wid = blockIdx.x;
  const int b = wid >> 11;
  const int dblk = (wid >> 8) & 7;
  const int hblk = (wid >> 4) & 15;
  const int wblk = wid & 15;

  // async-stage window input tile (12288 B) into LDS, contiguous [64][96]
  async_stage(winh + (size_t)wid * 6144, sWin, 6, tid);
  __syncthreads();

  // ---- Q/K/V: wave computes row-block mt; A frags loaded once; B pipelined ----
  {
    h16 wa0 = loadA(sWin, mt * 16, 0, 96, lane);
    h16 wa1 = loadA(sWin, mt * 16, 32, 96, lane);
    h16 wa2 = loadA(sWin, mt * 16, 64, 96, lane);

    h16 b0 = loadB(wqt, 96, 0, 0, lane);
    h16 b1 = loadB(wqt, 96, 0, 32, lane);
    h16 b2 = loadB(wqt, 96, 0, 64, lane);
#pragma unroll
    for (int nt = 0; nt < 6; nt++) {  // Q
      h16 c0, c1, c2;
      if (nt < 5) {
        c0 = loadB(wqt, 96, (nt + 1) * 16, 0, lane);
        c1 = loadB(wqt, 96, (nt + 1) * 16, 32, lane);
        c2 = loadB(wqt, 96, (nt + 1) * 16, 64, lane);
      } else {  // prefetch first K tile instead
        c0 = loadB(wkt, 96, 0, 0, lane);
        c1 = loadB(wkt, 96, 0, 32, lane);
        c2 = loadB(wkt, 96, 0, 64, lane);
      }
      f8 acc = wmma16(wa0, b0, f8zero());
      acc = wmma16(wa1, b1, acc);
      acc = wmma16(wa2, b2, acc);
      float bias_n = bq[nt * 16 + nidx];
#pragma unroll
      for (int i = 0; i < 8; i++)
        sQ[(mt * 16 + mrow + i) * 104 + nt * 16 + nidx] = (_Float16)(acc[i] + bias_n);
      b0 = c0; b1 = c1; b2 = c2;
    }
#pragma unroll
    for (int nt = 0; nt < 6; nt++) {  // K
      h16 c0, c1, c2;
      if (nt < 5) {
        c0 = loadB(wkt, 96, (nt + 1) * 16, 0, lane);
        c1 = loadB(wkt, 96, (nt + 1) * 16, 32, lane);
        c2 = loadB(wkt, 96, (nt + 1) * 16, 64, lane);
      } else {  // prefetch first V tile
        c0 = loadB(wvt, 96, 0, 0, lane);
        c1 = loadB(wvt, 96, 0, 32, lane);
        c2 = loadB(wvt, 96, 0, 64, lane);
      }
      f8 acc = wmma16(wa0, b0, f8zero());
      acc = wmma16(wa1, b1, acc);
      acc = wmma16(wa2, b2, acc);
      float bias_n = bk[nt * 16 + nidx];
#pragma unroll
      for (int i = 0; i < 8; i++)
        sK[(mt * 16 + mrow + i) * 104 + nt * 16 + nidx] = (_Float16)(acc[i] + bias_n);
      b0 = c0; b1 = c1; b2 = c2;
    }
#pragma unroll
    for (int nt = 0; nt < 6; nt++) {  // V (transposed; one b128 store per lane)
      h16 c0, c1, c2;
      if (nt < 5) {
        c0 = loadB(wvt, 96, (nt + 1) * 16, 0, lane);
        c1 = loadB(wvt, 96, (nt + 1) * 16, 32, lane);
        c2 = loadB(wvt, 96, (nt + 1) * 16, 64, lane);
      }
      f8 acc = wmma16(wa0, b0, f8zero());
      acc = wmma16(wa1, b1, acc);
      acc = wmma16(wa2, b2, acc);
      float bias_n = bv[nt * 16 + nidx];
      h8 pk;
#pragma unroll
      for (int i = 0; i < 8; i++) pk[i] = (_Float16)(acc[i] + bias_n);
      *(h8*)(sVt + (nt * 16 + nidx) * 72 + mt * 16 + mrow) = pk;
      b0 = c0; b1 = c1; b2 = c2;
    }
  }
  __syncthreads();

  const float scale = 0.20412414523f;  // 1/sqrt(24)
  for (int h = 0; h < 4; h++) {
    // scores[64,64] = Q_h[64,24] @ K_h^T (K padded 24->32 with zeros)
    {
      h16 qa = loadA_k24(sQ, mt * 16, h * 24, 104, lane);
#pragma unroll
      for (int nt = 0; nt < 4; nt++) {
        f8 acc = wmma16(qa, loadB_k24(sK, 104, nt * 16, h * 24, lane), f8zero());
#pragma unroll
        for (int i = 0; i < 8; i++)
          sS[(mt * 16 + mrow + i) * 72 + nt * 16 + nidx] =
              (_Float16)(acc[i] * scale);
      }
    }
    __syncthreads();
    // softmax rows, in place (f32 math, f16 storage)
    if (tid < 64) {
      _Float16* row = sS + tid * 72;
      float mx = -1e30f;
      for (int j = 0; j < 64; j++) mx = fmaxf(mx, (float)row[j]);
      float s = 0.f;
      for (int j = 0; j < 64; j++) {
        float e = __expf((float)row[j] - mx);
        s += e;
        row[j] = (_Float16)e;
      }
      float inv = 1.f / s;
      for (int j = 0; j < 64; j++) row[j] = (_Float16)((float)row[j] * inv);
    }
    __syncthreads();
    // O_h[64,24] = attn[64,64] @ V_h[64,24]
    {
      h16 pa0 = loadA(sS, mt * 16, 0, 72, lane);
      h16 pa1 = loadA(sS, mt * 16, 32, 72, lane);
      const _Float16* vtB = sVt + (size_t)(h * 24) * 72;
#pragma unroll
      for (int nt = 0; nt < 2; nt++) {
        f8 acc = wmma16(pa0, loadB(vtB, 72, nt * 16, 0, lane), f8zero());
        acc = wmma16(pa1, loadB(vtB, 72, nt * 16, 32, lane), acc);
        int n = nt * 16 + nidx;
        if (n < 24) {
#pragma unroll
          for (int i = 0; i < 8; i++)
            sO[(mt * 16 + mrow + i) * 104 + h * 24 + n] = (_Float16)acc[i];
        }
      }
    }
    __syncthreads();
  }

  // ---- out-proj [64,96]x[96,96] + residual(x) -> x2, stage rows for LN2 ----
  {
    h16 oa0 = loadA(sO, mt * 16, 0, 104, lane);
    h16 oa1 = loadA(sO, mt * 16, 32, 104, lane);
    h16 oa2 = loadA(sO, mt * 16, 64, 104, lane);
    h16 b0 = loadB(wot, 96, 0, 0, lane);
    h16 b1 = loadB(wot, 96, 0, 32, lane);
    h16 b2 = loadB(wot, 96, 0, 64, lane);
#pragma unroll
    for (int nt = 0; nt < 6; nt++) {
      h16 c0, c1, c2;
      if (nt < 5) {
        c0 = loadB(wot, 96, (nt + 1) * 16, 0, lane);
        c1 = loadB(wot, 96, (nt + 1) * 16, 32, lane);
        c2 = loadB(wot, 96, (nt + 1) * 16, 64, lane);
      }
      f8 acc = wmma16(oa0, b0, f8zero());
      acc = wmma16(oa1, b1, acc);
      acc = wmma16(oa2, b2, acc);
      int n = nt * 16 + nidx;
      float bo_n = bo[n];
#pragma unroll
      for (int i = 0; i < 8; i++) {
        int l = mt * 16 + mrow + i;
        int dd = l >> 4, hh = (l >> 2) & 3, ww = l & 3;
        size_t g =
            ((((size_t)(b * 32 + dblk * 4 + dd) * 64 + (hblk * 4 + hh)) * 64) +
             (wblk * 4 + ww)) * 96 + n;
        float v = x[g] + acc[i] + bo_n;
        x2[g] = v;
        sX2[l * 100 + n] = v;
      }
      b0 = c0; b1 = c1; b2 = c2;
    }
  }
  __syncthreads();

  // ---- fused LayerNorm2 (each window row is a full C=96 token) ----
  if (tid < 64) {
    int l = tid;
    float mu = 0.f;
    for (int c = 0; c < 96; c++) mu += sX2[l * 100 + c];
    mu *= (1.f / 96.f);
    float var = 0.f;
    for (int c = 0; c < 96; c++) {
      float d = sX2[l * 100 + c] - mu;
      var += d * d;
    }
    var *= (1.f / 96.f);
    float rs = rsqrtf(var + 1e-5f);
    int dd = l >> 4, hh = (l >> 2) & 3, ww = l & 3;
    size_t tok = (((size_t)(b * 32 + dblk * 4 + dd) * 64 + (hblk * 4 + hh)) * 64 +
                  (wblk * 4 + ww));
    _Float16* dst = ln2h + tok * 96;
#pragma unroll
    for (int cc = 0; cc < 12; cc++) {
      h8 pk;
#pragma unroll
      for (int j = 0; j < 8; j++) {
        int c = cc * 8 + j;
        pk[j] = (_Float16)((sX2[l * 100 + c] - mu) * rs * ln2g[c] + ln2b[c]);
      }
      *(h8*)(dst + cc * 8) = pk;
    }
  }
}

// ---------------------------------------------------------------------------
// Kernel 3: MLP: out = x2 + gelu(ln2h @ W1 + b1) @ W2 + b2
// One 128-thread block per 64 tokens; 64x384 hidden kept entirely in LDS f16.
// ---------------------------------------------------------------------------
__global__ __launch_bounds__(128) void mlp_kernel(
    const _Float16* __restrict__ ln2h, const float* __restrict__ x2,
    const _Float16* __restrict__ w1t, const float* __restrict__ b1,
    const _Float16* __restrict__ w2t, const float* __restrict__ b2,
    float* __restrict__ out) {
  __shared__ __align__(16) _Float16 sA[64 * 96];    // 12288 B contiguous
  __shared__ __align__(16) _Float16 sH[64 * 392];   // 50176 B
  const int tid = threadIdx.x;
  const int lane = tid & 31;
  const int mt = __builtin_amdgcn_readfirstlane(tid >> 5);  // scalar wave id
  const int nidx = lane & 15;
  const int mrow = (lane >> 4) * 8;
  const size_t tb = (size_t)blockIdx.x * 64;

  async_stage(ln2h + tb * 96, sA, 6, tid);
  __syncthreads();

  // H = gelu(A @ W1^T + b1): wave owns row-block mt; 24 N-tiles, B pipelined
  {
    h16 aa0 = loadA(sA, mt * 16, 0, 96, lane);
    h16 aa1 = loadA(sA, mt * 16, 32, 96, lane);
    h16 aa2 = loadA(sA, mt * 16, 64, 96, lane);
    h16 b0 = loadB(w1t, 96, 0, 0, lane);
    h16 b1f = loadB(w1t, 96, 0, 32, lane);
    h16 b2f = loadB(w1t, 96, 0, 64, lane);
#pragma unroll
    for (int nt = 0; nt < 24; nt++) {
      h16 c0, c1, c2;
      if (nt < 23) {
        c0 = loadB(w1t, 96, (nt + 1) * 16, 0, lane);
        c1 = loadB(w1t, 96, (nt + 1) * 16, 32, lane);
        c2 = loadB(w1t, 96, (nt + 1) * 16, 64, lane);
      }
      f8 acc = wmma16(aa0, b0, f8zero());
      acc = wmma16(aa1, b1f, acc);
      acc = wmma16(aa2, b2f, acc);
      float b1_n = b1[nt * 16 + nidx];
#pragma unroll
      for (int i = 0; i < 8; i++) {
        float v = acc[i] + b1_n;
        v = 0.5f * v * (1.0f + erff(v * 0.70710678118f));  // exact gelu
        sH[(mt * 16 + mrow + i) * 392 + nt * 16 + nidx] = (_Float16)v;
      }
      b0 = c0; b1f = c1; b2f = c2;
    }
  }
  __syncthreads();

  // out = H @ W2^T + b2 + x2 : preload all 12 A fragments, pipeline B over ks
  {
    h16 ha[12];
#pragma unroll
    for (int ks = 0; ks < 12; ks++)
      ha[ks] = loadA(sH, mt * 16, ks * 32, 392, lane);
#pragma unroll
    for (int nt = 0; nt < 6; nt++) {
      h16 bb = loadB(w2t, 384, nt * 16, 0, lane);
      f8 acc = f8zero();
#pragma unroll
      for (int ks = 0; ks < 12; ks++) {
        h16 nb;
        if (ks < 11) nb = loadB(w2t, 384, nt * 16, (ks + 1) * 32, lane);
        acc = wmma16(ha[ks], bb, acc);
        bb = nb;
      }
      int n = nt * 16 + nidx;
      float b2_n = b2[n];
#pragma unroll
      for (int i = 0; i < 8; i++) {
        int m = mt * 16 + mrow + i;
        size_t g = (tb + m) * 96 + n;
        out[g] = x2[g] + acc[i] + b2_n;
      }
    }
  }
}

// ---------------------------------------------------------------------------
extern "C" void kernel_launch(void* const* d_in, const int* in_sizes, int n_in,
                              void* d_out, int out_size, void* d_ws, size_t ws_size,
                              hipStream_t stream) {
  const float* x = (const float*)d_in[0];
  const float* ln1g = (const float*)d_in[1];
  const float* ln1b = (const float*)d_in[2];
  const float* wq = (const float*)d_in[3];
  const float* bq = (const float*)d_in[4];
  const float* wk = (const float*)d_in[5];
  const float* bk = (const float*)d_in[6];
  const float* wv = (const float*)d_in[7];
  const float* bv = (const float*)d_in[8];
  const float* wo = (const float*)d_in[9];
  const float* bo = (const float*)d_in[10];
  const float* ln2g = (const float*)d_in[11];
  const float* ln2b = (const float*)d_in[12];
  const float* w1 = (const float*)d_in[13];
  const float* b1 = (const float*)d_in[14];
  const float* w2 = (const float*)d_in[15];
  const float* b2 = (const float*)d_in[16];
  float* out = (float*)d_out;

  char* ws = (char*)d_ws;
  size_t off = 0;
  _Float16* winh = (_Float16*)(ws + off); off += 50331648;   // [4096][64][96] f16
  _Float16* ln2h = (_Float16*)(ws + off); off += 50331648;   // [262144][96]  f16
  float* x2 = (float*)(ws + off);         off += 100663296;  // [262144][96]  f32
  _Float16* wqt = (_Float16*)(ws + off);  off += 18432;
  _Float16* wkt = (_Float16*)(ws + off);  off += 18432;
  _Float16* wvt = (_Float16*)(ws + off);  off += 18432;
  _Float16* wot = (_Float16*)(ws + off);  off += 18432;
  _Float16* w1t = (_Float16*)(ws + off);  off += 73728;      // [384][96] f16
  _Float16* w2t = (_Float16*)(ws + off);  off += 73728;      // [96][384] f16

  prep_kernel<<<144, 256, 0, stream>>>(wq, wk, wv, wo, w1, w2, wqt, wkt, wvt, wot,
                                       w1t, w2t);
  ln1_kernel<<<2048, 128, 0, stream>>>(x, ln1g, ln1b, winh);
  attn_kernel<<<4096, 128, 0, stream>>>(x, winh, wqt, bq, wkt, bk, wvt, bv, wot, bo,
                                        ln2g, ln2b, x2, ln2h);
  mlp_kernel<<<4096, 128, 0, stream>>>(ln2h, x2, w1t, b1, w2t, b2, out);
}
